// BiRGAT_27049704030448
// MI455X (gfx1250) — matrix operand order repaired and checked by reading, not compiled, not run on hardware
//
#include <hip/hip_runtime.h>
#include <hip/hip_bf16.h>
#include <cstdint>

// ---------------------------------------------------------------------------
// Types for CDNA5 WMMA (wave32): 16x16x32 bf16 -> f32
// ---------------------------------------------------------------------------
typedef __bf16 bf16_t;
typedef __attribute__((ext_vector_type(16))) __bf16 v16bf;
typedef __attribute__((ext_vector_type(8)))  __bf16 v8bf;
typedef __attribute__((ext_vector_type(8)))  float  v8f;

// Order-preserving float<->uint key for atomic segment-max
__device__ __forceinline__ unsigned fkey(float f) {
  unsigned u = __float_as_uint(f);
  return (u >> 31) ? ~u : (u | 0x80000000u);
}
__device__ __forceinline__ float funkey(unsigned k) {
  unsigned u = (k & 0x80000000u) ? (k & 0x7FFFFFFFu) : ~k;
  return __uint_as_float(u);
}

// ---------------------------------------------------------------------------
// GEMM: C[M,N] = act(A[M,K] @ W[K,N] + bias[N]),  act: 0=none, 1=relu
// Block = 128 threads (4 waves).  Block tile = 32 rows x 64 cols.
// Wave w owns N-tile w (16 cols) and both 16-row M-tiles (2 WMMAs / K-step,
// sharing one B fragment).  A tile staged in LDS as bf16; fragments read as
// contiguous 16-byte chunks per the ISA 7.12.2 layout.
// REQUIRES: M % 32 == 0 (all call sites use M = 4000 or 20000).
// ---------------------------------------------------------------------------
__global__ void wmma_gemm(const float* __restrict__ A, const float* __restrict__ W,
                          const float* __restrict__ bias, float* __restrict__ C,
                          int M, int K, int N, int act)
{
  __shared__ __bf16 As[32 * 32];          // 2 KB

  const int tid  = threadIdx.x;
  const int wave = tid >> 5;
  const int lane = tid & 31;
  const int hl   = lane >> 4;             // lane half
  const int l16  = lane & 15;

  const int m0 = blockIdx.x * 32;
  const int n0 = blockIdx.y * 64 + wave * 16;

  const int bn  = n0 + l16;
  const int bnc = (bn < N) ? bn : (N - 1);   // clamped column for loads
  const bool bn_ok = bn < N;

  // A-staging assignment: 4 threads per row, 8 consecutive k each (float4 x2)
  const int srow = tid >> 2;              // 0..31
  const int scol = (tid & 3) * 8;         // 0,8,16,24

  v8f acc0 = {}, acc1 = {};

  auto kstep = [&](int k0, bool tail) {
    // ---- stage A tile (32 x 32 fp32 -> bf16 in LDS), coalesced b128 loads ----
    {
      const float* srcrow = A + (size_t)(m0 + srow) * K;
      float t0, t1, t2, t3, t4, t5, t6, t7;
      if (!tail) {
        const float4 f0 = *(const float4*)(srcrow + k0 + scol);
        const float4 f1 = *(const float4*)(srcrow + k0 + scol + 4);
        t0 = f0.x; t1 = f0.y; t2 = f0.z; t3 = f0.w;
        t4 = f1.x; t5 = f1.y; t6 = f1.z; t7 = f1.w;
      } else {
        float tv[8];
#pragma unroll
        for (int i = 0; i < 8; ++i) {
          int k  = k0 + scol + i;
          int kc = (k < K) ? k : (K - 1);
          float v = srcrow[kc];            // unconditional (clamped) load
          tv[i] = (k < K) ? v : 0.f;       // value select, no exec branch
        }
        t0 = tv[0]; t1 = tv[1]; t2 = tv[2]; t3 = tv[3];
        t4 = tv[4]; t5 = tv[5]; t6 = tv[6]; t7 = tv[7];
      }
      __bf16* dst = &As[srow * 32 + scol];
      dst[0] = (__bf16)t0; dst[1] = (__bf16)t1; dst[2] = (__bf16)t2; dst[3] = (__bf16)t3;
      dst[4] = (__bf16)t4; dst[5] = (__bf16)t5; dst[6] = (__bf16)t6; dst[7] = (__bf16)t7;
    }
    __syncthreads();

    // ---- B fragment: lane holds column bn, k = k0 + 8*hl + j (+8 for j >= 8)  ----
    v16bf b;
#pragma unroll
    for (int j = 0; j < 16; ++j) {
      int k = k0 + 8 * hl + j + ((j >= 8) ? 8 : 0);
      float v;
      if (!tail) {
        v = W[(size_t)k * N + bnc];
      } else {
        int kc = (k < K) ? k : (K - 1);
        float w = W[(size_t)kc * N + bnc];
        v = (k < K) ? w : 0.f;
      }
      b[j] = (__bf16)v;
    }

    // ---- A fragments from LDS: two contiguous v8bf chunks per tile ----
    const v8bf* Ap = (const v8bf*)As;      // 4 chunks per row
    v8bf a0lo = Ap[l16 * 4 + hl];
    v8bf a0hi = Ap[l16 * 4 + 2 + hl];
    v8bf a1lo = Ap[(16 + l16) * 4 + hl];
    v8bf a1hi = Ap[(16 + l16) * 4 + 2 + hl];
    v16bf a0 = __builtin_shufflevector(a0lo, a0hi, 0,1,2,3,4,5,6,7,8,9,10,11,12,13,14,15);
    v16bf a1 = __builtin_shufflevector(a1lo, a1hi, 0,1,2,3,4,5,6,7,8,9,10,11,12,13,14,15);

    acc0 = __builtin_amdgcn_wmma_f32_16x16x32_bf16(false, a0, false, b,
                                                   (short)0, acc0, false, false);
    acc1 = __builtin_amdgcn_wmma_f32_16x16x32_bf16(false, a1, false, b,
                                                   (short)0, acc1, false, false);
    __syncthreads();
  };

  int k0 = 0;
  for (; k0 + 32 <= K; k0 += 32) kstep(k0, false);
  if (k0 < K) kstep(k0, true);             // uniform tail branch (K=2000 only)

  // ---- store: C/D layout lane half -> M base 0/8, VGPR j -> row j ----
  if (bn_ok) {
    float bv = bias ? bias[bn] : 0.f;
#pragma unroll
    for (int j = 0; j < 8; ++j) {
      int r = j + 8 * hl;
      float v0 = acc0[j] + bv;
      if (act == 1) v0 = fmaxf(v0, 0.f);
      C[(size_t)(m0 + r) * N + bn] = v0;
      float v1 = acc1[j] + bv;
      if (act == 1) v1 = fmaxf(v1, 0.f);
      C[(size_t)(m0 + 16 + r) * N + bn] = v1;
    }
  }
}

// ---------------------------------------------------------------------------
// GATv2 edge passes.  eidx layout: [2,E] row-major (row0=src, row1=dst).
// Thread per (edge, head); e/ee stored [E,H].
// ---------------------------------------------------------------------------
__global__ void edge_score_max(const float* __restrict__ xl, const float* __restrict__ xr,
                               const int* __restrict__ eidx, const float* __restrict__ att,
                               float* __restrict__ e, unsigned* __restrict__ emaxk,
                               int E, int H, int C)
{
  int t = blockIdx.x * blockDim.x + threadIdx.x;
  if (t >= E * H) return;
  int edge = t / H, h = t % H;
  int s = eidx[edge], d = eidx[E + edge];
  const float* pl = xl + (size_t)s * H * C + (size_t)h * C;
  const float* pr = xr + (size_t)d * H * C + (size_t)h * C;
  const float* pa = att + (size_t)h * C;
  float acc = 0.f;
  for (int c = 0; c < C; ++c) {
    float m = pl[c] + pr[c];
    m = m > 0.f ? m : 0.2f * m;            // leaky_relu(0.2)
    acc += m * pa[c];
  }
  e[t] = acc;
  atomicMax(&emaxk[(size_t)d * H + h], fkey(acc));
}

__global__ void edge_softmax_denom(const int* __restrict__ eidx,
                                   const unsigned* __restrict__ emaxk,
                                   float* __restrict__ e, float* __restrict__ denom,
                                   int E, int H)
{
  int t = blockIdx.x * blockDim.x + threadIdx.x;
  if (t >= E * H) return;
  int edge = t / H, h = t % H;
  int d = eidx[E + edge];
  float ex = __expf(e[t] - funkey(emaxk[(size_t)d * H + h]));
  e[t] = ex;
  atomicAdd(&denom[(size_t)d * H + h], ex);
}

__global__ void edge_aggregate(const float* __restrict__ xl, const int* __restrict__ eidx,
                               const float* __restrict__ ee, const float* __restrict__ denom,
                               float* __restrict__ agg, int E, int H, int C)
{
  int t = blockIdx.x * blockDim.x + threadIdx.x;
  if (t >= E * H) return;
  int edge = t / H, h = t % H;
  int s = eidx[edge], d = eidx[E + edge];
  float alpha = ee[t] / (denom[(size_t)d * H + h] + 1e-16f);
  const float* pl = xl + (size_t)s * H * C + (size_t)h * C;
  float* pd = agg + (size_t)d * H * C + (size_t)h * C;
  for (int c = 0; c < C; ++c)
    atomicAdd(&pd[c], pl[c] * alpha);
}

// new = elu(agg + rel_bias(es) + tile(self_lin, H))     (concat layers)
__global__ void combine_concat(const float* __restrict__ agg, const float* __restrict__ sl,
                               const float* __restrict__ b1, const float* __restrict__ b2,
                               float* __restrict__ out, int Nn, int HC, int outc)
{
  int t = blockIdx.x * blockDim.x + threadIdx.x;
  if (t >= Nn * HC) return;
  int i = t / HC, j = t % HC;
  float v = agg[t] + b1[j] + (b2 ? b2[j] : 0.f) + sl[(size_t)i * outc + (j % outc)];
  out[t] = v > 0.f ? v : (__expf(v) - 1.f);
}

// new = elu(mean_h(agg) + rel_bias(es) + self_lin)       (final GAT layer)
__global__ void combine_mean(const float* __restrict__ agg, const float* __restrict__ sl,
                             const float* __restrict__ b1, const float* __restrict__ b2,
                             float* __restrict__ out, int Nn, int H, int C)
{
  int t = blockIdx.x * blockDim.x + threadIdx.x;
  if (t >= Nn * C) return;
  int i = t / C, j = t % C;
  float s = 0.f;
  for (int h = 0; h < H; ++h) s += agg[(size_t)i * H * C + (size_t)h * C + j];
  float v = s / (float)H + b1[j] + (b2 ? b2[j] : 0.f) + sl[(size_t)i * C + j];
  out[t] = v > 0.f ? v : (__expf(v) - 1.f);
}

__global__ void concat2(const float* __restrict__ a, const float* __restrict__ b,
                        float* __restrict__ out, int Nn, int Ca, int Cb)
{
  int t = blockIdx.x * blockDim.x + threadIdx.x;
  int Ct = Ca + Cb;
  if (t >= Nn * Ct) return;
  int i = t / Ct, j = t % Ct;
  out[t] = (j < Ca) ? a[(size_t)i * Ca + j] : b[(size_t)i * Cb + (j - Ca)];
}

// ---------------------------------------------------------------------------
// Host orchestration
// ---------------------------------------------------------------------------
static void gemm(hipStream_t st, const float* A, const float* W, const float* b,
                 float* C, int M, int K, int N, int act)
{
  dim3 grid(M / 32, (N + 63) / 64);        // M is always a multiple of 32 here
  wmma_gemm<<<grid, 128, 0, st>>>(A, W, b, C, M, K, N, act);
}

extern "C" void kernel_launch(void* const* d_in, const int* in_sizes, int n_in,
                              void* d_out, int out_size, void* d_ws, size_t ws_size,
                              hipStream_t stream)
{
  (void)n_in; (void)out_size; (void)ws_size;
  auto F = [&](int i) { return (const float*)d_in[i]; };
  auto I = [&](int i) { return (const int*)d_in[i]; };
  const int E = in_sizes[3] / 2;
  const int H = 4;

  // ---- workspace carve (floats) ----
  float* p = (float*)d_ws;
  auto alloc = [&](size_t n) { float* r = p; p += n; return r; };
  float* projm = alloc((size_t)4000 * 256);
  float* projc = alloc((size_t)4000 * 256);
  float* xg[2] = { alloc((size_t)20000 * 128), alloc((size_t)20000 * 128) };
  float* xm[2] = { alloc((size_t)4000 * 128),  alloc((size_t)4000 * 128) };
  float* xc[2] = { alloc((size_t)4000 * 128),  alloc((size_t)4000 * 128) };
  float* xl    = alloc((size_t)20000 * 256);
  float* xr    = alloc((size_t)20000 * 256);
  float* ebuf  = alloc((size_t)E * H);
  unsigned* emaxk = (unsigned*)alloc((size_t)20000 * H);
  float* denom = alloc((size_t)20000 * H);
  float* agg[3] = { alloc((size_t)4000 * 256), alloc((size_t)4000 * 256),
                    alloc((size_t)20000 * 256) };                 // mrna, cna, gene
  float* slbuf = alloc((size_t)20000 * 64);
  float* feat  = alloc((size_t)4000 * 128);
  float* h1    = alloc((size_t)4000 * 128);

  // ---- omics projections: relu(x @ W + b) ----
  gemm(stream, F(0), F(7),  F(8),  projm, 4000, 2000, 256, 1);
  gemm(stream, F(1), F(9),  F(10), projc, 4000, 2000, 256, 1);

  const int counts[3] = {4000, 4000, 20000};       // mrna, cna, gene
  const int relS[4] = {0, 2, 1, 2};                // RELS src types
  const int relD[4] = {2, 0, 2, 1};                // RELS dst types
  const int relE[4] = {3, 4, 5, 6};                // edge input indices

  const float* xin[3];
  float* xout[3];

  for (int L = 0; L < 3; ++L) {
    const int din    = (L == 0) ? 256 : 128;
    const int outc   = (L == 2) ? 64 : 32;
    const int concat = (L != 2);
    const int HC     = H * outc;
    const int base   = 11 + 30 * L;                // first rel-param index
    const int slb    = 35 + 30 * L;                // first self-lin param index

    if (L == 0)      { xin[0] = projm; xin[1] = projc; xin[2] = F(2);
                       xout[0] = xm[0]; xout[1] = xc[0]; xout[2] = xg[0]; }
    else if (L == 1) { xin[0] = xm[0]; xin[1] = xc[0]; xin[2] = xg[0];
                       xout[0] = xm[1]; xout[1] = xc[1]; xout[2] = xg[1]; }
    else             { xin[0] = xm[1]; xin[1] = xc[1]; xin[2] = xg[1];
                       xout[0] = xm[0]; xout[1] = xc[0]; xout[2] = xg[0]; }

    for (int t = 0; t < 3; ++t)
      hipMemsetAsync(agg[t], 0, (size_t)counts[t] * HC * sizeof(float), stream);

    for (int r = 0; r < 4; ++r) {
      const int s = relS[r], d = relD[r];
      const int rb = base + 6 * r;                 // Wl,bl,Wr,br,att,bias
      gemm(stream, xin[s], F(rb + 0), F(rb + 1), xl, counts[s], din, HC, 0);
      gemm(stream, xin[d], F(rb + 2), F(rb + 3), xr, counts[d], din, HC, 0);
      hipMemsetAsync(emaxk, 0, (size_t)counts[d] * H * sizeof(unsigned), stream);
      hipMemsetAsync(denom, 0, (size_t)counts[d] * H * sizeof(float), stream);
      const int nt = E * H, blks = (nt + 255) / 256;
      edge_score_max<<<blks, 256, 0, stream>>>(xl, xr, I(relE[r]), F(rb + 4),
                                               ebuf, emaxk, E, H, outc);
      edge_softmax_denom<<<blks, 256, 0, stream>>>(I(relE[r]), emaxk, ebuf, denom, E, H);
      edge_aggregate<<<blks, 256, 0, stream>>>(xl, I(relE[r]), ebuf, denom,
                                               agg[d], E, H, outc);
    }

    // relation biases feeding each dst node type
    const float* bdst1[3] = { F(base + 6 * 1 + 5),   // mrna  <- gene->mrna
                              F(base + 6 * 3 + 5),   // cna   <- gene->cna
                              F(base + 6 * 0 + 5) }; // gene  <- mrna->gene
    const float* bdst2[3] = { nullptr, nullptr,
                              F(base + 6 * 2 + 5) }; // gene  <- cna->gene

    for (int t = 0; t < 3; ++t) {
      gemm(stream, xin[t], F(slb + 2 * t), F(slb + 2 * t + 1), slbuf,
           counts[t], din, outc, 0);
      if (concat) {
        int nt = counts[t] * HC;
        combine_concat<<<(nt + 255) / 256, 256, 0, stream>>>(
            agg[t], slbuf, bdst1[t], bdst2[t], xout[t], counts[t], HC, outc);
      } else {
        int nt = counts[t] * outc;
        combine_mean<<<(nt + 255) / 256, 256, 0, stream>>>(
            agg[t], slbuf, bdst1[t], bdst2[t], xout[t], counts[t], H, outc);
      }
    }
  }

  // ---- integration MLP: concat(mrna, cna) -> relu(lin1) -> lin2 ----
  concat2<<<(4000 * 128 + 255) / 256, 256, 0, stream>>>(xm[0], xc[0], feat, 4000, 64, 64);
  gemm(stream, feat, F(101), F(102), h1, 4000, 128, 128, 1);
  gemm(stream, h1,   F(103), F(104), (float*)d_out, 4000, 128, 5, 0);
}
